// KNRM_16647293239572
// MI455X (gfx1250) — matrix operand order repaired and checked by reading, not compiled, not run on hardware
//
#include <hip/hip_runtime.h>
#include <hip/hip_bf16.h>

typedef __attribute__((ext_vector_type(16))) _Float16 v16h;
typedef __attribute__((ext_vector_type(8)))  float    v8f;
typedef __attribute__((ext_vector_type(4)))  _Float16 h4;

#define QROW  136   // halves per LDS embedding row (128 + 8 pad -> 272B stride)
#define MMROW 68    // floats per mm row (64 + 4 pad -> 272B stride)
#define NKER  21

// e^{10-k} for k=1..19 (ratio constants of adjacent Gaussian kernels, sigma=0.1)
__device__ __constant__ float CKAP[19] = {
    8103.0839f, 2980.9580f, 1096.6332f, 403.42879f, 148.41316f,
    54.598150f, 20.085537f, 7.3890561f, 2.7182818f, 1.0f,
    0.36787944f, 0.13533528f, 0.049787068f, 0.018315639f, 0.0067379470f,
    0.0024787522f, 0.00091188197f, 0.00033546263f, 0.00012340980f
};

__global__ __launch_bounds__(256) void knrm_feats_kernel(
    const int* __restrict__ q_ids1, const int* __restrict__ d_ids1,
    const int* __restrict__ q_ids2, const int* __restrict__ d_ids2,
    const float* __restrict__ E, float* __restrict__ feats)
{
    __shared__ _Float16 qs [64 * QROW];   // 17408 B  normalized query tile (f16)
    __shared__ _Float16 dsb[64 * QROW];   // 17408 B  normalized doc tile (f16)
    __shared__ float    mmb[64 * MMROW];  // 17408 B  matching-matrix block (f32)
    __shared__ float    Sb [64 * NKER];   //  5376 B  per-(q-row,kernel) sums

    const int tid  = threadIdx.x;
    const int wave = tid >> 5;
    const int lane = tid & 31;
    const int b    = blockIdx.x;
    const int pair = blockIdx.y;
    const int* qids = pair ? q_ids2 : q_ids1;
    const int* dids = pair ? d_ids2 : d_ids1;

    for (int i = tid; i < 64 * NKER; i += 256) Sb[i] = 0.f;

    // ---- gather + L2-normalize 64 query rows -> f16 in LDS ----
    for (int i = 0; i < 8; ++i) {
        const int row = wave + i * 8;
        const int id  = qids[b * 64 + row];
        const float4 ev = *(const float4*)(E + (size_t)id * 128 + lane * 4);
        float ss = ev.x*ev.x + ev.y*ev.y + ev.z*ev.z + ev.w*ev.w;
        #pragma unroll
        for (int off = 16; off > 0; off >>= 1) ss += __shfl_xor(ss, off, 32);
        // 1/max(sqrt(ss),1e-12) == rsqrt(max(ss,1e-24)) for ss >= 0
        const float scale = rsqrtf(fmaxf(ss, 1e-24f));
        h4 hv;
        hv.x = (_Float16)(ev.x * scale);
        hv.y = (_Float16)(ev.y * scale);
        hv.z = (_Float16)(ev.z * scale);
        hv.w = (_Float16)(ev.w * scale);
        *(h4*)(qs + row * QROW + lane * 4) = hv;
    }
    __syncthreads();

    // ---- preload A fragments (this wave's fixed 16-row m-tile, all 4 K-steps)
    // ISA 16-bit A 16x32 layout: lane m holds K {0..7,16..23} (lanes 0-15) or
    // {8..15,24..31} (lanes 16-31), shifted by k0.
    const int m0   = (wave & 3) * 16;
    const int hs   = lane >> 4;
    const int lrow = lane & 15;
    v16h afrag[4];
    #pragma unroll
    for (int kk = 0; kk < 4; ++kk) {
        const _Float16* p = qs + (m0 + lrow) * QROW + kk * 32 + hs * 8;
        union { uint4 u[2]; v16h h; } t;
        t.u[0] = *(const uint4*)p;
        t.u[1] = *(const uint4*)(p + 16);
        afrag[kk] = t.h;
    }

    // ---- per-thread kernel-pool accumulators (q-row pm, column quarter pq)
    float acc[NKER];
    #pragma unroll
    for (int k = 0; k < NKER; ++k) acc[k] = 0.f;
    const int pm = tid >> 2;
    const int pq = tid & 3;

    for (int blk = 0; blk < 16; ++blk) {
        // -- gather + normalize 64 document rows of this block --
        for (int i = 0; i < 8; ++i) {
            const int row = wave + i * 8;
            const int id  = dids[b * 1024 + blk * 64 + row];
            const float4 ev = *(const float4*)(E + (size_t)id * 128 + lane * 4);
            float ss = ev.x*ev.x + ev.y*ev.y + ev.z*ev.z + ev.w*ev.w;
            #pragma unroll
            for (int off = 16; off > 0; off >>= 1) ss += __shfl_xor(ss, off, 32);
            const float scale = rsqrtf(fmaxf(ss, 1e-24f));
            h4 hv;
            hv.x = (_Float16)(ev.x * scale);
            hv.y = (_Float16)(ev.y * scale);
            hv.z = (_Float16)(ev.z * scale);
            hv.w = (_Float16)(ev.w * scale);
            *(h4*)(dsb + row * QROW + lane * 4) = hv;
        }
        __syncthreads();

        // -- WMMA: mm block [64 x 64] = q[64x128] . d_blk[64x128]^T --
        // B 32x16 layout: lane holds column n = lane%16, K {0..7,16..23} (lanes 0-15)
        // or {8..15,24..31} (lanes 16-31), shifted by k0 (same as A).
        #pragma unroll
        for (int j = 0; j < 2; ++j) {
            const int n0 = ((wave >> 2) * 2 + j) * 16;
            v8f c = {};
            #pragma unroll
            for (int kk = 0; kk < 4; ++kk) {
                const _Float16* p = dsb + (n0 + lrow) * QROW + kk * 32 + hs * 8;
                union { uint4 u[2]; v16h h; } t;
                t.u[0] = *(const uint4*)p;
                t.u[1] = *(const uint4*)(p + 16);
                c = __builtin_amdgcn_wmma_f32_16x16x32_f16(
                        false, afrag[kk], false, t.h, (short)0, c, false, false);
            }
            // C layout: lane, vgpr r -> (M = r + 8*(lane/16), N = lane%16)
            const int cm = m0 + 8 * hs;
            #pragma unroll
            for (int r = 0; r < 8; ++r)
                mmb[(cm + r) * MMROW + n0 + lrow] = c[r];
        }
        __syncthreads();

        // -- kernel pooling via exp-recurrence: r_k = r_{k-1} * e^{10x} * e^{10-k}
        const float* rowp = mmb + pm * MMROW + pq * 16;
        #pragma unroll
        for (int jj = 0; jj < 4; ++jj) {
            const float4 xv = *(const float4*)(rowp + jj * 4);
            const float xs[4] = { xv.x, xv.y, xv.z, xv.w };
            #pragma unroll
            for (int e = 0; e < 4; ++e) {
                const float x  = xs[e];
                const float d0 = x + 0.95f;
                float r = __expf(-50.f * d0 * d0);   // kernel 0 (mu=-0.95)
                const float u = __expf(10.f * x);
                acc[0] += r;
                #pragma unroll
                for (int k = 1; k < 20; ++k) {
                    r = r * u * CKAP[k - 1];
                    acc[k] += r;
                }
                const float t1 = x - 1.f;            // kernel 20 (mu=1, sigma=1e-3)
                acc[20] += __expf(-500000.f * t1 * t1);
            }
        }
        __syncthreads();
    }

    // ---- fold the 4 column-quarter threads of each q-row into Sb ----
    #pragma unroll
    for (int k = 0; k < NKER; ++k) atomicAdd(&Sb[pm * NKER + k], acc[k]);
    __syncthreads();

    // ---- feats[k] = sum_m log1p(S[m][k]) ----
    if (tid < NKER) {
        float f = 0.f;
        for (int m = 0; m < 64; ++m) f += __logf(1.f + Sb[m * NKER + tid]);
        feats[((size_t)pair * 512 + b) * NKER + tid] = f;
    }
}

__global__ void knrm_mlp_kernel(const float* __restrict__ feats,
    const float* __restrict__ W1, const float* __restrict__ b1,
    const float* __restrict__ W2, const float* __restrict__ b2,
    const float* __restrict__ W3, const float* __restrict__ b3,
    float* __restrict__ out)
{
    const int bb = blockIdx.x * blockDim.x + threadIdx.x;
    if (bb >= 512) return;
    float logit[2];
    #pragma unroll
    for (int p = 0; p < 2; ++p) {
        const float* f = feats + ((size_t)p * 512 + bb) * NKER;
        float h1[10];
        #pragma unroll
        for (int j = 0; j < 10; ++j) h1[j] = b1[j];
        for (int i = 0; i < NKER; ++i) {
            const float fi = f[i];
            #pragma unroll
            for (int j = 0; j < 10; ++j) h1[j] = fmaf(fi, W1[i * 10 + j], h1[j]);
        }
        #pragma unroll
        for (int j = 0; j < 10; ++j) h1[j] = fmaxf(h1[j], 0.f);
        float h2[5];
        #pragma unroll
        for (int j = 0; j < 5; ++j) h2[j] = b2[j];
        #pragma unroll
        for (int i = 0; i < 10; ++i)
            #pragma unroll
            for (int j = 0; j < 5; ++j) h2[j] = fmaf(h1[i], W2[i * 5 + j], h2[j]);
        float z = b3[0];
        #pragma unroll
        for (int i = 0; i < 5; ++i) z = fmaf(fmaxf(h2[i], 0.f), W3[i], z);
        logit[p] = z;
    }
    out[bb] = 1.f / (1.f + __expf(-(logit[0] - logit[1])));
}

extern "C" void kernel_launch(void* const* d_in, const int* in_sizes, int n_in,
                              void* d_out, int out_size, void* d_ws, size_t ws_size,
                              hipStream_t stream) {
    const int*   q1 = (const int*)  d_in[0];
    const int*   d1 = (const int*)  d_in[1];
    const int*   q2 = (const int*)  d_in[2];
    const int*   d2 = (const int*)  d_in[3];
    const float* E  = (const float*)d_in[4];
    const float* W1 = (const float*)d_in[5];
    const float* b1 = (const float*)d_in[6];
    const float* W2 = (const float*)d_in[7];
    const float* b2 = (const float*)d_in[8];
    const float* W3 = (const float*)d_in[9];
    const float* b3 = (const float*)d_in[10];

    float* feats = (float*)d_ws;   // 2 * 512 * 21 floats of scratch

    dim3 grid(512, 2), block(256);
    knrm_feats_kernel<<<grid, block, 0, stream>>>(q1, d1, q2, d2, E, feats);
    knrm_mlp_kernel<<<2, 256, 0, stream>>>(feats, W1, b1, W2, b2, W3, b3,
                                           (float*)d_out);
}
